// formerblock_43928925503951
// MI455X (gfx1250) — compile-verified, matrix operand order checked by reading, not run on hardware
//
#include <hip/hip_runtime.h>
#include <hip/hip_bf16.h>

typedef __attribute__((ext_vector_type(16))) _Float16 v16h;
typedef __attribute__((ext_vector_type(8)))  float    v8f;

#define WPB 4  // windows per block

// WMMA 16-bit A/B fragment K-pattern: lane L holds 16 halves; element e maps to
// k = kbase + (e&7) + ((e>>3)<<4), kbase = (L&16)?8:0.  (ISA 7.12.2)
__device__ __forceinline__ int kpat(int e, int lane) {
  return (((lane & 16) >> 1)) + (e & 7) + ((e >> 3) << 4);
}

static __device__ __forceinline__ v8f wmma16(v16h a, v16h b, v8f c) {
  return __builtin_amdgcn_wmma_f32_16x16x32_f16(false, a, false, b, (short)0, c,
                                                false, false);
}

// ---------------------------------------------------------------------------
// Kernel 1: repack Wq/Wk/Wv (fp32 -> f16, Wq pre-scaled by d^-0.5) into
// A-fragment-contiguous layout, plus combined [rel_w | rel_h] B-fragments.
// ws layout (halves): [3 weights][16 mtiles][8 ktiles][32 lanes][16 elems]
// then at offset 196608: [2 ktiles][32 lanes][16 elems] for rel B-matrix.
// ---------------------------------------------------------------------------
__global__ void pack_weights_kernel(const float* __restrict__ Wq,
                                    const float* __restrict__ Wk,
                                    const float* __restrict__ Wv,
                                    const float* __restrict__ rel_h,
                                    const float* __restrict__ rel_w,
                                    _Float16* __restrict__ ws) {
  int lane = threadIdx.x & 31;
  int fid = blockIdx.x * (blockDim.x >> 5) + (threadIdx.x >> 5);
  if (fid >= 386) return;
  if (fid < 384) {
    int w  = fid >> 7;         // 0..2
    int mt = (fid >> 3) & 15;  // 0..15
    int kt = fid & 7;          // 0..7
    const float* W = (w == 0) ? Wq : (w == 1) ? Wk : Wv;
    float scale = (w == 0) ? 0.125f : 1.0f;  // d^-0.5, d = 64
    int m = lane & 15;
    _Float16* dst = ws + fid * 512 + lane * 16;
#pragma unroll
    for (int e = 0; e < 16; ++e) {
      int k = kpat(e, lane);
      dst[e] = (_Float16)(W[(mt * 16 + m) * 256 + kt * 32 + k] * scale);
    }
  } else {
    // rel B-matrix: B(k=d-channel, n=col) ; cols 0..6 = rel_w[n], 7..13 = rel_h[n-7]
    int kt = fid - 384;
    int j = lane & 15;
    _Float16* dst = ws + 196608 + kt * 512 + lane * 16;
#pragma unroll
    for (int e = 0; e < 16; ++e) {
      int kd = kt * 32 + kpat(e, lane);
      float v = 0.0f;
      if (j < 7)       v = rel_w[j * 64 + kd];
      else if (j < 14) v = rel_h[(j - 7) * 64 + kd];
      dst[e] = (_Float16)v;
    }
  }
}

// ---------------------------------------------------------------------------
// Kernel 2: the fused window-MHSA. One WG = 256 threads (8 wave32), WPB windows.
// ---------------------------------------------------------------------------
__launch_bounds__(256)
__global__ void former_mhsa_kernel(const float* __restrict__ x,
                                   const _Float16* __restrict__ wpk,
                                   float* __restrict__ out) {
  extern __shared__ _Float16 sm[];
  // LDS map (in halves):
  const int XF  = 0;       // X B-fragments: [win][kt(8)][lane(32)][16] = 16384
  const int QKV = 16384;   // [wgt*WPB+win][c(256)][t(16)] = 49152
  const int PS  = 65536;   // softmax scratch: [wave(8)][m(16)][n(16)] = 2048
  const int tid = threadIdx.x, lane = tid & 31, wave = tid >> 5;
  const long winBase = (long)blockIdx.x * WPB;

  // ---- Phase 1: stage X (fp32, contiguous [c][t] per window) into LDS as
  // f16 pre-swizzled WMMA B-fragments. Coalesced float4 global reads.
  {
    const float4* xg = (const float4*)(x + winBase * 4096);
#pragma unroll
    for (int i = 0; i < 16; ++i) {
      float4 v = xg[i * 256 + tid];
      int f0 = (i * 256 + tid) * 4;
      float vv[4] = {v.x, v.y, v.z, v.w};
#pragma unroll
      for (int j = 0; j < 4; ++j) {
        int f   = f0 + j;
        int win = f >> 12, rem = f & 4095;
        int c = rem >> 4, t = rem & 15;
        int kt = c >> 5, k = c & 31;
        int dl = t + (((k >> 3) & 1) << 4);       // fragment lane
        int e  = (k & 7) | ((k & 16) >> 1);       // fragment element
        sm[XF + ((win * 8 + kt) * 32 + dl) * 16 + e] = (_Float16)vv[j];
      }
    }
  }
  __syncthreads();

  // ---- Phase 2: Q/K/V projections. job = (weight, mtile); A-fragments held
  // in registers, reused across WPB windows.
  for (int job = wave; job < 48; job += 8) {
    int w = job / 16, mt = job % 16;
    v16h a[8];
    const v16h* ab = (const v16h*)wpk + (w * 16 + mt) * 8 * 32;
#pragma unroll
    for (int kt = 0; kt < 8; ++kt) a[kt] = ab[kt * 32 + lane];  // 2x b128 each
#pragma unroll
    for (int win = 0; win < WPB; ++win) {
      v8f acc = {};
#pragma unroll
      for (int kt = 0; kt < 8; ++kt) {
        v16h b = *(const v16h*)(sm + XF + ((win * 8 + kt) * 32 + lane) * 16);
        acc = wmma16(a[kt], b, acc);
      }
      // C layout: vgpr r -> row mt*16 + 8*(lane>=16) + r, col = lane&15
      int mrow = mt * 16 + ((lane >> 4) << 3);
      int n = lane & 15;
      _Float16* q = sm + QKV + ((w * WPB + win) * 256 + mrow) * 16 + n;
#pragma unroll
      for (int r = 0; r < 8; ++r) q[r * 16] = (_Float16)acc[r];
    }
  }
  __syncthreads();

  // ---- Phase 3: attention. pair = (window, head); 2 pairs per wave.
  const v16h* rf = (const v16h*)(wpk + 196608);
  for (int p = wave * 2; p < wave * 2 + 2; ++p) {
    int win = p >> 2, head = p & 3;
    const _Float16* Q = sm + QKV + ((0 * WPB + win) * 256 + head * 64) * 16;
    const _Float16* K = sm + QKV + ((1 * WPB + win) * 256 + head * 64) * 16;
    const _Float16* V = sm + QKV + ((2 * WPB + win) * 256 + head * 64) * 16;
    int n = lane & 15;

    // q as A (m=token,k=d): transposed gather; k as B (k=d,n=token): natural.
    v16h qa[2], kb[2];
#pragma unroll
    for (int kt = 0; kt < 2; ++kt) {
#pragma unroll
      for (int e = 0; e < 16; ++e) {
        int kd = kt * 32 + kpat(e, lane);
        qa[kt][e] = Q[kd * 16 + n];
        kb[kt][e] = K[kd * 16 + n];
      }
    }
    v8f lg = {}, P = {};
    lg = wmma16(qa[0], kb[0], lg);
    lg = wmma16(qa[1], kb[1], lg);
    P  = wmma16(qa[0], rf[lane], P);        // cols 0..6: q.rel_w ; 7..13: q.rel_h
    P  = wmma16(qa[1], rf[32 + lane], P);

    // rel bias: logits[m][n] += P[m][wn-wm+3] + P[m][7 + hn-hm+3]
    int hb = lane & 16;
    int mr0 = (lane >> 4) << 3;
#pragma unroll
    for (int r = 0; r < 8; ++r) {
      int m = mr0 + r;
      int c1 = (n & 3) - (m & 3) + 3;
      int c2 = 7 + (n >> 2) - (m >> 2) + 3;
      lg[r] += __shfl(P[r], hb | c1, 32) + __shfl(P[r], hb | c2, 32);
    }

    // softmax over the 16 key columns (per 16-lane half), store probs as f16
    _Float16* Ps = sm + PS + wave * 256;
#pragma unroll
    for (int r = 0; r < 8; ++r) {
      float xv = lg[r];
      float mx = xv;
#pragma unroll
      for (int off = 8; off; off >>= 1) mx = fmaxf(mx, __shfl_xor(mx, off, 32));
      float ev = __expf(xv - mx);
      float s = ev;
#pragma unroll
      for (int off = 8; off; off >>= 1) s += __shfl_xor(s, off, 32);
      Ps[(mr0 + r) * 16 + n] = (_Float16)(ev / s);
    }

    // out = P(16x16, K padded to 32 with zeros) @ V(16x64), 4 N-tiles of 16
    v16h wa;
#pragma unroll
    for (int e = 0; e < 16; ++e) {
      int k = kpat(e, lane);
      wa[e] = (k < 16) ? Ps[n * 16 + k] : (_Float16)0.0f;  // A: m=n token row
    }
    float* og = out + (winBase + win) * 4096 + head * 64 * 16;
#pragma unroll
    for (int dt = 0; dt < 4; ++dt) {
      v16h vb;
#pragma unroll
      for (int e = 0; e < 16; ++e) {
        int k = kpat(e, lane);
        vb[e] = (k < 16) ? V[(dt * 16 + n) * 16 + k] : (_Float16)0.0f;
      }
      v8f oc = {};
      oc = wmma16(wa, vb, oc);
#pragma unroll
      for (int r = 0; r < 8; ++r)
        og[(dt * 16 + n) * 16 + (mr0 + r)] = oc[r];
    }
  }
}

extern "C" void kernel_launch(void* const* d_in, const int* in_sizes, int n_in,
                              void* d_out, int out_size, void* d_ws, size_t ws_size,
                              hipStream_t stream) {
  const float* x     = (const float*)d_in[0];  // [32,256,64,64]
  const float* Wq    = (const float*)d_in[1];  // [256,256]
  const float* Wk    = (const float*)d_in[2];
  const float* Wv    = (const float*)d_in[3];
  const float* rel_h = (const float*)d_in[4];  // [7,64]
  const float* rel_w = (const float*)d_in[5];  // [7,64]
  float* out = (float*)d_out;
  _Float16* ws = (_Float16*)d_ws;

  // 386 fragments, 8 per 256-thread block
  pack_weights_kernel<<<49, 256, 0, stream>>>(Wq, Wk, Wv, rel_h, rel_w, ws);

  const int shmem = (16384 + 49152 + 2048) * 2;  // 135168 bytes of LDS
  hipFuncSetAttribute((const void*)former_mhsa_kernel,
                      hipFuncAttributeMaxDynamicSharedMemorySize, shmem);
  former_mhsa_kernel<<<8192 / WPB, 256, shmem, stream>>>(x, ws, out);
}
